// DiMBlock_70635032150692
// MI455X (gfx1250) — matrix operand order repaired and checked by reading, not compiled, not run on hardware
//
#include <hip/hip_runtime.h>
#include <math.h>

// ---------------------------------------------------------------------------
// Problem constants (DiM block: B=4, L=2048, DIM=1024)
// ---------------------------------------------------------------------------
#define BATCH   4
#define SEQL    2048
#define DIM     1024
#define DINNER  2048
#define DSTATE  16
#define DCONV   4
#define DTRANK  64
#define MLPHID  4096
#define MROWS   (BATCH * SEQL)      // 8192
#define NXZ     (2 * DINNER)        // 4096
#define XDBLW   (DTRANK + 2*DSTATE) // 96
#define MODW    (6 * DIM)           // 6144

// ---------------------------------------------------------------------------
// Types for CDNA5 WMMA (wave32)
// ---------------------------------------------------------------------------
typedef __attribute__((ext_vector_type(16))) __bf16 v16bf;
typedef __attribute__((ext_vector_type(8)))  float  v8f;

union Frag { uint4 u[2]; v16bf v; };

__device__ __forceinline__ unsigned short f2bf(float f) {
    unsigned int u = __float_as_uint(f);
    unsigned int r = (u + 0x7FFFu + ((u >> 16) & 1u)) >> 16;   // RNE
    return (unsigned short)r;
}
__device__ __forceinline__ float siluf(float x)  { return x / (1.f + __expf(-x)); }
__device__ __forceinline__ float softplusf(float x) { return (x > 20.f) ? x : log1pf(__expf(x)); }
__device__ __forceinline__ float gelu_tanh(float x) {
    float x3 = x * x * x;
    float t = tanhf(0.7978845608028654f * (x + 0.044715f * x3));
    return 0.5f * x * (1.f + t);
}

// gfx1250 async global->LDS copy (ASYNCcnt-tracked), inline asm for toolchain
// portability (builtin arity differs between ROCm 7.2 and amdgpu-toolchain).
__device__ __forceinline__ void async_ld_b128(unsigned lds_off, unsigned long long gaddr) {
    asm volatile("global_load_async_to_lds_b128 %0, %1, off"
                 :: "v"(lds_off), "v"(gaddr) : "memory");
}
__device__ __forceinline__ void wait_async0() {
    asm volatile("s_wait_asynccnt 0x0" ::: "memory");
}

// ---------------------------------------------------------------------------
// Generic bf16 WMMA GEMM:  C[M,N](f32) = A[M,K](bf16, row stride lda)
//                                      x W[N,K](bf16, row stride ldw)^T
// Block tile 128x128, BK=64, 256 threads = 8 waves (4x2), wave tile 32x64.
// Double-buffered LDS fed by GLOBAL_LOAD_ASYNC_TO_LDS_B128; fused epilogues.
// ---------------------------------------------------------------------------
#define GBM 128
#define GBN 128
#define GBK 64
#define GLDT (GBK + 8)   // padded LDS row stride in bf16 (144 B, 16B aligned)

// Issue one 128x64 bf16 tile as 4 async b128 loads per thread.
// OOB rows are clamped (async loads cannot zero-fill); epilogue bounds-checks.
__device__ __forceinline__ void async_tile(unsigned lds_base,
                                           const unsigned short* __restrict__ g,
                                           int ld, int maxRow, int tileRow,
                                           int k0, int tid)
{
#pragma unroll
    for (int t = 0; t < 4; ++t) {
        int idx = tid + t * 256;
        int row = idx >> 3;              // 8 x b128 per row
        int c8  = idx & 7;
        int gr  = tileRow + row;
        if (gr >= maxRow) gr = maxRow - 1;
        unsigned long long ga = (unsigned long long)(g + (size_t)gr * ld + k0 + c8 * 8);
        unsigned la = lds_base + (unsigned)(row * (GLDT * 2) + c8 * 16);
        async_ld_b128(la, ga);
    }
}

// mode: 0 none | 1 +bias | 2 softplus(+bias) | 3 gelu(+bias)
//       4 gate[b,n]*acc | 5 addend + gate[b,n]*(acc+bias)
__global__ __launch_bounds__(256)
void gemm_bf16_wmma(const unsigned short* __restrict__ A, int lda,
                    const unsigned short* __restrict__ W, int ldw,
                    float* __restrict__ Cf, unsigned short* __restrict__ Cbf, int ldc,
                    int M, int N, int K,
                    const float* __restrict__ bias,
                    const float* __restrict__ gate, int gate_stride,
                    const float* __restrict__ addend,
                    int rows_per_batch, int mode)
{
    __shared__ unsigned short As[2][GBM * GLDT];
    __shared__ unsigned short Bs[2][GBN * GLDT];

    const int tid  = threadIdx.x;
    const int lane = tid & 31;
    const int wave = tid >> 5;
    const int wm   = wave & 3;       // 0..3  -> 32-row band
    const int wn   = wave >> 2;      // 0..1  -> 64-col band
    const int h    = lane >> 4;      // lane half
    const int l16  = lane & 15;

    const int tileM = blockIdx.y * GBM;
    const int tileN = blockIdx.x * GBN;

    unsigned asBase[2], bsBase[2];
    asBase[0] = (unsigned)(unsigned long long)(const void*)&As[0][0];
    asBase[1] = (unsigned)(unsigned long long)(const void*)&As[1][0];
    bsBase[0] = (unsigned)(unsigned long long)(const void*)&Bs[0][0];
    bsBase[1] = (unsigned)(unsigned long long)(const void*)&Bs[1][0];

    v8f acc[2][4];
    const v8f vzero = {0.f,0.f,0.f,0.f,0.f,0.f,0.f,0.f};
#pragma unroll
    for (int mi = 0; mi < 2; ++mi)
#pragma unroll
        for (int ni = 0; ni < 4; ++ni) acc[mi][ni] = vzero;

    const int nTiles = K / GBK;

    // ---- prologue: stage tile 0 into buffer 0 ----
    async_tile(asBase[0], A, lda, M, tileM, 0, tid);
    async_tile(bsBase[0], W, ldw, N, tileN, 0, tid);
    wait_async0();
    __syncthreads();

    for (int it = 0; it < nTiles; ++it) {
        const int cur = it & 1, nxt = cur ^ 1;
        const int k0  = it * GBK;

        // ---- issue async loads for next tile while computing this one ----
        if (it + 1 < nTiles) {
            async_tile(asBase[nxt], A, lda, M, tileM, k0 + GBK, tid);
            async_tile(bsBase[nxt], W, ldw, N, tileN, k0 + GBK, tid);
        }
        // warm L2 two tiles ahead (global_prefetch_b8)
        if (it + 2 < nTiles) {
            int row = tid >> 1, c8 = tid & 1;
            __builtin_prefetch(A + (size_t)(tileM + row) * lda + (k0 + 2*GBK) + c8 * 32, 0, 3);
            __builtin_prefetch(W + (size_t)(tileN + row) * ldw + (k0 + 2*GBK) + c8 * 32, 0, 3);
        }

        // ---- two WMMA K-steps (K=32 each) on the current LDS buffer ----
#pragma unroll
        for (int ks = 0; ks < 2; ++ks) {
            v16bf afrag[2], bfrag[4];
#pragma unroll
            for (int mi = 0; mi < 2; ++mi) {
                int row = wm * 32 + mi * 16 + l16;
                const unsigned short* p = &As[cur][row * GLDT + ks * 32 + h * 8];
                Frag f;
                f.u[0] = *(const uint4*)(p);        // K = base + h*8 .. +7
                f.u[1] = *(const uint4*)(p + 16);   // K = base + 16 + h*8 ..
                afrag[mi] = f.v;
            }
#pragma unroll
            for (int ni = 0; ni < 4; ++ni) {
                int nr = wn * 64 + ni * 16 + l16;
                const unsigned short* p = &Bs[cur][nr * GLDT + ks * 32 + h * 8];
                Frag f;
                f.u[0] = *(const uint4*)(p);        // K = base + h*8 .. +7
                f.u[1] = *(const uint4*)(p + 16);   // K = base + 16 + h*8 ..
                bfrag[ni] = f.v;
            }
#pragma unroll
            for (int mi = 0; mi < 2; ++mi)
#pragma unroll
                for (int ni = 0; ni < 4; ++ni)
                    acc[mi][ni] = __builtin_amdgcn_wmma_f32_16x16x32_bf16(
                        /*neg_a=*/false, afrag[mi], /*neg_b=*/false, bfrag[ni],
                        /*c_mod=*/(short)0, acc[mi][ni],
                        /*reuse_a=*/false, /*reuse_b=*/false);
        }

        // our async writes into `nxt` done, then workgroup-wide visibility
        wait_async0();
        __syncthreads();
    }

    // ---- epilogue ----
#pragma unroll
    for (int mi = 0; mi < 2; ++mi) {
#pragma unroll
        for (int ni = 0; ni < 4; ++ni) {
            int gc = tileN + wn * 64 + ni * 16 + l16;
            if (gc >= N) continue;
            float bv = bias ? bias[gc] : 0.f;
#pragma unroll
            for (int r = 0; r < 8; ++r) {
                int gr = tileM + wm * 32 + mi * 16 + h * 8 + r;
                if (gr >= M) continue;
                float v = acc[mi][ni][r];
                if (mode == 1)      v += bv;
                else if (mode == 2) v = softplusf(v + bv);
                else if (mode == 3) v = gelu_tanh(v + bv);
                else if (mode == 4) {
                    int b = gr / rows_per_batch;
                    v *= gate[b * gate_stride + gc];
                } else if (mode == 5) {
                    int b = gr / rows_per_batch;
                    v = addend[(size_t)gr * ldc + gc] + gate[b * gate_stride + gc] * (v + bv);
                }
                if (Cf)  Cf [(size_t)gr * ldc + gc] = v;
                if (Cbf) Cbf[(size_t)gr * ldc + gc] = f2bf(v);
            }
        }
    }
}

// ---------------------------------------------------------------------------
// LayerNorm + adaLN modulate -> bf16 (one 256-thread block per row)
// ---------------------------------------------------------------------------
__global__ __launch_bounds__(256)
void ln_modulate(const float* __restrict__ X,
                 const float* __restrict__ w, const float* __restrict__ b,
                 const float* __restrict__ shift, const float* __restrict__ scale,
                 int mod_stride, unsigned short* __restrict__ Ybf,
                 int D, int rows_per_batch, float eps)
{
    __shared__ float red[256];
    const int tid = threadIdx.x;
    const int row = blockIdx.x;
    const int bi  = row / rows_per_batch;
    const float* x = X + (size_t)row * D;

    float s = 0.f;
    for (int i = tid; i < D; i += 256) s += x[i];
    red[tid] = s; __syncthreads();
    for (int off = 128; off > 0; off >>= 1) {
        if (tid < off) red[tid] += red[tid + off];
        __syncthreads();
    }
    float mu = red[0] * (1.f / D);
    __syncthreads();

    float vs = 0.f;
    for (int i = tid; i < D; i += 256) { float d = x[i] - mu; vs += d * d; }
    red[tid] = vs; __syncthreads();
    for (int off = 128; off > 0; off >>= 1) {
        if (tid < off) red[tid] += red[tid + off];
        __syncthreads();
    }
    float rstd = rsqrtf(red[0] * (1.f / D) + eps);

    for (int i = tid; i < D; i += 256) {
        float y = (x[i] - mu) * rstd;
        if (w) y = y * w[i] + b[i];
        float sc = scale[bi * mod_stride + i];
        float sh = shift[bi * mod_stride + i];
        y = y * (1.f + sc) + sh;
        Ybf[(size_t)row * D + i] = f2bf(y);
    }
}

// ---------------------------------------------------------------------------
// Causal depthwise conv (D_CONV=4) + bias + SiLU ; dual f32 + bf16 output
// ---------------------------------------------------------------------------
__global__ __launch_bounds__(256)
void conv_silu(const float* __restrict__ xz, const float* __restrict__ cw,
               const float* __restrict__ cb,
               float* __restrict__ xc, unsigned short* __restrict__ xcbf)
{
    size_t gid = (size_t)blockIdx.x * 256 + threadIdx.x;   // [B,L,Di]
    int d = (int)(gid % DINNER);
    size_t rt = gid / DINNER;
    int t = (int)(rt % SEQL);
    size_t bL = rt - t;                                     // b*L
    float acc = cb[d];
#pragma unroll
    for (int j = 0; j < DCONV; ++j) {
        int tt = t - (DCONV - 1) + j;
        if (tt >= 0) acc += cw[d * DCONV + j] * xz[(bL + tt) * (size_t)NXZ + d];
    }
    acc = siluf(acc);
    xc[gid]   = acc;
    xcbf[gid] = f2bf(acc);
}

// ---------------------------------------------------------------------------
// Selective scan: one lane per (b, d) channel, 16-state recurrence in VGPRs,
// B_t/C_t staged in LDS in 64-timestep chunks. Fuses +x*D, *silu(z), ->bf16.
// ---------------------------------------------------------------------------
#define TCHUNK 64
__global__ __launch_bounds__(256)
void selective_scan(const float* __restrict__ dtb, const float* __restrict__ xc,
                    const float* __restrict__ xdbl, const float* __restrict__ xz,
                    const float* __restrict__ A_log, const float* __restrict__ Dp,
                    unsigned short* __restrict__ ybf)
{
    __shared__ float bc[TCHUNK * 32];
    const int b = blockIdx.x >> 3;                    // Di/256 = 8 blocks per batch
    const int d = ((blockIdx.x & 7) << 8) + threadIdx.x;

    float a[DSTATE], hst[DSTATE];
#pragma unroll
    for (int n = 0; n < DSTATE; ++n) {
        a[n] = -__expf(A_log[d * DSTATE + n]);
        hst[n] = 0.f;
    }
    const float dp = Dp[d];
    const size_t rowbase = (size_t)b * SEQL;

    for (int t0 = 0; t0 < SEQL; t0 += TCHUNK) {
#pragma unroll
        for (int j = 0; j < (TCHUNK * 32) / 256; ++j) {
            int idx = threadIdx.x + j * 256;
            int tl = idx >> 5, c = idx & 31;
            bc[idx] = xdbl[(rowbase + t0 + tl) * XDBLW + DTRANK + c];
        }
        __syncthreads();
        for (int tl = 0; tl < TCHUNK; ++tl) {
            size_t r   = rowbase + t0 + tl;
            float dtv  = dtb[r * DINNER + d];
            float xv   = xc [r * DINNER + d];
            float dtx  = dtv * xv;
            const float* Bt = &bc[tl * 32];
            float y = 0.f;
#pragma unroll
            for (int n = 0; n < DSTATE; ++n) {
                float dA = __expf(dtv * a[n]);
                hst[n] = hst[n] * dA + dtx * Bt[n];
                y += hst[n] * Bt[DSTATE + n];
            }
            y += xv * dp;
            float zv = xz[r * (size_t)NXZ + DINNER + d];
            ybf[r * DINNER + d] = f2bf(y * siluf(zv));
        }
        __syncthreads();
    }
}

// ---------------------------------------------------------------------------
// Small helpers: silu(cond), adaLN GEMV, f32->bf16 convert, f32 copy
// ---------------------------------------------------------------------------
__global__ void silu_vec(const float* __restrict__ in, float* __restrict__ out, int n) {
    int i = blockIdx.x * 256 + threadIdx.x;
    if (i < n) out[i] = siluf(in[i]);
}

__global__ __launch_bounds__(256)
void ada_gemv(const float* __restrict__ sc, const float* __restrict__ w,
              const float* __restrict__ bias, float* __restrict__ mods)
{
    int gid = blockIdx.x * 256 + threadIdx.x;       // BATCH * MODW
    int b = gid / MODW, j = gid % MODW;
    const float4* wp = (const float4*)(w + (size_t)j * DIM);
    const float4* xp = (const float4*)(sc + (size_t)b * DIM);
    float s = 0.f;
#pragma unroll 4
    for (int k = 0; k < DIM / 4; ++k) {
        float4 A = xp[k], Wv = wp[k];
        s += A.x * Wv.x + A.y * Wv.y + A.z * Wv.z + A.w * Wv.w;
    }
    mods[gid] = s + bias[j];
}

__global__ void f32_to_bf16_vec(const float* __restrict__ in,
                                unsigned short* __restrict__ out, size_t n) {
    size_t i = (size_t)blockIdx.x * 256 + threadIdx.x;
    size_t stride = (size_t)gridDim.x * 256;
    for (; i < n; i += stride) out[i] = f2bf(in[i]);
}

__global__ void copy_f32(const float* __restrict__ in, float* __restrict__ out, size_t n) {
    size_t i = (size_t)blockIdx.x * 256 + threadIdx.x;
    size_t stride = (size_t)gridDim.x * 256;
    for (; i < n; i += stride) out[i] = in[i];
}

// ---------------------------------------------------------------------------
// Host-side orchestration
// ---------------------------------------------------------------------------
extern "C" void kernel_launch(void* const* d_in, const int* in_sizes, int n_in,
                              void* d_out, int out_size, void* d_ws, size_t ws_size,
                              hipStream_t stream)
{
    const float* hidden    = (const float*)d_in[0];
    const float* cond      = (const float*)d_in[1];
    const float* norm_w    = (const float*)d_in[2];
    const float* norm_b    = (const float*)d_in[3];
    const float* ada_w     = (const float*)d_in[4];
    const float* ada_b     = (const float*)d_in[5];
    const float* in_proj_w = (const float*)d_in[6];
    const float* conv_w    = (const float*)d_in[7];
    const float* conv_b    = (const float*)d_in[8];
    const float* x_proj_w  = (const float*)d_in[9];
    const float* dt_proj_w = (const float*)d_in[10];
    const float* dt_proj_b = (const float*)d_in[11];
    const float* A_log     = (const float*)d_in[12];
    const float* Dp        = (const float*)d_in[13];
    const float* out_proj_w= (const float*)d_in[14];
    const float* fc1_w     = (const float*)d_in[15];
    const float* fc1_b     = (const float*)d_in[16];
    const float* fc2_w     = (const float*)d_in[17];
    const float* fc2_b     = (const float*)d_in[18];

    float* out_hs  = (float*)d_out;                             // [B,L,DIM]
    float* out_res = (float*)d_out + (size_t)MROWS * DIM;       // residual

    // deterministic workspace carve-out
    char* wsp = (char*)d_ws;
    auto alloc = [&](size_t bytes) -> void* {
        void* p = (void*)wsp;
        wsp += (bytes + 255) & ~(size_t)255;
        return p;
    };
    float*          sc        = (float*)alloc((size_t)BATCH * DIM * 4);
    float*          mods      = (float*)alloc((size_t)BATCH * MODW * 4);
    unsigned short* wb_inproj = (unsigned short*)alloc((size_t)NXZ * DIM * 2);
    unsigned short* wb_xproj  = (unsigned short*)alloc((size_t)XDBLW * DINNER * 2);
    unsigned short* wb_dtproj = (unsigned short*)alloc((size_t)DINNER * DTRANK * 2);
    unsigned short* wb_outproj= (unsigned short*)alloc((size_t)DIM * DINNER * 2);
    unsigned short* wb_fc1    = (unsigned short*)alloc((size_t)MLPHID * DIM * 2);
    unsigned short* wb_fc2    = (unsigned short*)alloc((size_t)DIM * MLPHID * 2);
    unsigned short* xmod_bf   = (unsigned short*)alloc((size_t)MROWS * DIM * 2);
    float*          xzbuf     = (float*)alloc((size_t)MROWS * NXZ * 4);
    float*          xconv     = (float*)alloc((size_t)MROWS * DINNER * 4);
    unsigned short* xconv_bf  = (unsigned short*)alloc((size_t)MROWS * DINNER * 2);
    float*          xdbl      = (float*)alloc((size_t)MROWS * XDBLW * 4);
    unsigned short* xdbl_bf   = (unsigned short*)alloc((size_t)MROWS * XDBLW * 2);
    float*          dtbuf     = (float*)alloc((size_t)MROWS * DINNER * 4);
    unsigned short* ybf       = (unsigned short*)alloc((size_t)MROWS * DINNER * 2);
    float*          hsbuf     = (float*)alloc((size_t)MROWS * DIM * 4);
    unsigned short* h2mod_bf  = (unsigned short*)alloc((size_t)MROWS * DIM * 2);
    unsigned short* h1bf      = (unsigned short*)alloc((size_t)MROWS * MLPHID * 2);
    (void)ws_size; (void)in_sizes; (void)n_in; (void)out_size;

    // 1) adaLN: mods = silu(cond) @ ada_w^T + ada_b
    silu_vec<<<(BATCH * DIM + 255) / 256, 256, 0, stream>>>(cond, sc, BATCH * DIM);
    ada_gemv<<<(BATCH * MODW) / 256, 256, 0, stream>>>(sc, ada_w, ada_b, mods);
    const float* shift_msa = mods + 0 * DIM;
    const float* scale_msa = mods + 1 * DIM;
    const float* gate_msa  = mods + 2 * DIM;
    const float* shift_mlp = mods + 3 * DIM;
    const float* scale_mlp = mods + 4 * DIM;
    const float* gate_mlp  = mods + 5 * DIM;

    // 2) weight demotion f32 -> bf16 (consumed by the WMMA GEMMs)
    f32_to_bf16_vec<<<2048, 256, 0, stream>>>(in_proj_w,  wb_inproj,  (size_t)NXZ * DIM);
    f32_to_bf16_vec<<<512,  256, 0, stream>>>(x_proj_w,   wb_xproj,   (size_t)XDBLW * DINNER);
    f32_to_bf16_vec<<<512,  256, 0, stream>>>(dt_proj_w,  wb_dtproj,  (size_t)DINNER * DTRANK);
    f32_to_bf16_vec<<<1024, 256, 0, stream>>>(out_proj_w, wb_outproj, (size_t)DIM * DINNER);
    f32_to_bf16_vec<<<2048, 256, 0, stream>>>(fc1_w,      wb_fc1,     (size_t)MLPHID * DIM);
    f32_to_bf16_vec<<<2048, 256, 0, stream>>>(fc2_w,      wb_fc2,     (size_t)DIM * MLPHID);

    // 3) LN(hidden) + modulate(shift_msa, scale_msa) -> bf16
    ln_modulate<<<MROWS, 256, 0, stream>>>(hidden, norm_w, norm_b,
                                           shift_msa, scale_msa, MODW,
                                           xmod_bf, DIM, SEQL, 1e-5f);

    // 4) in_proj: [8192,1024] x [4096,1024]^T -> xz [8192,4096]
    {
        dim3 g(NXZ / GBN, MROWS / GBM);
        gemm_bf16_wmma<<<g, 256, 0, stream>>>(xmod_bf, DIM, wb_inproj, DIM,
                                              xzbuf, nullptr, NXZ,
                                              MROWS, NXZ, DIM,
                                              nullptr, nullptr, 0, nullptr, SEQL, 0);
    }

    // 5) causal depthwise conv + SiLU
    conv_silu<<<((size_t)MROWS * DINNER) / 256, 256, 0, stream>>>(xzbuf, conv_w, conv_b,
                                                                  xconv, xconv_bf);

    // 6) x_proj: [8192,2048] x [96,2048]^T -> x_dbl (dt|B|C), dual f32+bf16
    {
        dim3 g((XDBLW + GBN - 1) / GBN, MROWS / GBM);
        gemm_bf16_wmma<<<g, 256, 0, stream>>>(xconv_bf, DINNER, wb_xproj, DINNER,
                                              xdbl, xdbl_bf, XDBLW,
                                              MROWS, XDBLW, DINNER,
                                              nullptr, nullptr, 0, nullptr, SEQL, 0);
    }

    // 7) dt_proj + softplus: A = x_dbl cols[0:64) (lda=96), W [2048,64]
    {
        dim3 g(DINNER / GBN, MROWS / GBM);
        gemm_bf16_wmma<<<g, 256, 0, stream>>>(xdbl_bf, XDBLW, wb_dtproj, DTRANK,
                                              dtbuf, nullptr, DINNER,
                                              MROWS, DINNER, DTRANK,
                                              dt_proj_b, nullptr, 0, nullptr, SEQL, 2);
    }

    // 8) selective scan (+x*D, *silu(z)) -> ybf
    selective_scan<<<BATCH * (DINNER / 256), 256, 0, stream>>>(dtbuf, xconv, xdbl, xzbuf,
                                                               A_log, Dp, ybf);

    // 9) out_proj with fused gate_msa: hs = gate_msa * (y @ out_proj_w^T)
    {
        dim3 g(DIM / GBN, MROWS / GBM);
        gemm_bf16_wmma<<<g, 256, 0, stream>>>(ybf, DINNER, wb_outproj, DINNER,
                                              hsbuf, nullptr, DIM,
                                              MROWS, DIM, DINNER,
                                              nullptr, gate_msa, MODW, nullptr, SEQL, 4);
    }

    // 10) LN(hs, no affine, eps=1e-6) + modulate(shift_mlp, scale_mlp) -> bf16
    ln_modulate<<<MROWS, 256, 0, stream>>>(hsbuf, nullptr, nullptr,
                                           shift_mlp, scale_mlp, MODW,
                                           h2mod_bf, DIM, SEQL, 1e-6f);

    // 11) fc1 + gelu(tanh) -> bf16 (feeds fc2 directly)
    {
        dim3 g(MLPHID / GBN, MROWS / GBM);
        gemm_bf16_wmma<<<g, 256, 0, stream>>>(h2mod_bf, DIM, wb_fc1, DIM,
                                              nullptr, h1bf, MLPHID,
                                              MROWS, MLPHID, DIM,
                                              fc1_b, nullptr, 0, nullptr, SEQL, 3);
    }

    // 12) fc2 + fused residual: out = hs + gate_mlp * (h1 @ fc2_w^T + fc2_b)
    {
        dim3 g(DIM / GBN, MROWS / GBM);
        gemm_bf16_wmma<<<g, 256, 0, stream>>>(h1bf, MLPHID, wb_fc2, MLPHID,
                                              out_hs, nullptr, DIM,
                                              MROWS, DIM, MLPHID,
                                              fc2_b, gate_mlp, MODW, hsbuf, SEQL, 5);
    }

    // 13) residual passthrough
    copy_f32<<<4096, 256, 0, stream>>>(hidden, out_res, (size_t)MROWS * DIM);
}
